// MultiHeadAttention_16028817949171
// MI455X (gfx1250) — matrix operand order repaired and checked by reading, not compiled, not run on hardware
//
#include <hip/hip_runtime.h>
#include <hip/hip_bf16.h>

typedef __bf16 bf16;
typedef __attribute__((ext_vector_type(16))) __bf16 v16bf;
typedef __attribute__((ext_vector_type(4)))  __bf16 v4bf;
typedef __attribute__((ext_vector_type(8)))  float  v8f;

#define WMMA_BF16(a, b, c) \
  __builtin_amdgcn_wmma_f32_16x16x32_bf16(false, (a), false, (b), (short)0, (c), false, false)

// Async DMA global -> LDS, 16B per lane (ISA §10 VGLOBAL async; tracked by ASYNCcnt)
__device__ __forceinline__ void async_copy_b128(void* lds_ptr, const void* gptr) {
  unsigned lds_off = (unsigned)(size_t)lds_ptr;  // flat shared addr[31:0] == LDS byte offset
  unsigned long long ga = (unsigned long long)(size_t)gptr;
  asm volatile("global_load_async_to_lds_b128 %0, %1, off"
               :: "v"(lds_off), "v"(ga) : "memory");
}
#define WAIT_ASYNCCNT(n) asm volatile("s_wait_asynccnt " #n ::: "memory")

constexpr int   B_   = 2;
constexpr int   T_   = 2048;
constexpr int   DIM_ = 1024;
constexpr int   H_   = 16;
constexpr int   HD_  = 64;
constexpr float SCALE_ = 0.03125f;  // 1/sqrt(1024)
constexpr int   LDA_ = 40;          // LDS row stride in bf16 (32 data + 8 pad); 80B, 16B-aligned

// ---------------------------------------------------------------------------
// f32 -> bf16 conversion (one-shot; halves GEMM global traffic, enables
// direct async-DMA staging since the TDM/async path cannot convert in flight)
// ---------------------------------------------------------------------------
__global__ void __launch_bounds__(256)
cvt_f32_bf16(const float* __restrict__ src, bf16* __restrict__ dst) {
  size_t i = ((size_t)blockIdx.x * 256 + threadIdx.x) * 4;
  float4 v = *(const float4*)(src + i);
  v4bf o = {(bf16)v.x, (bf16)v.y, (bf16)v.z, (bf16)v.w};
  *(v4bf*)(dst + i) = o;
}

// ---------------------------------------------------------------------------
// GEMM: Out = X @ W^T, all-bf16 inputs.  Block tile 128x128, 8 waves (4Mx2N),
// each wave 32x64 via 2x4 WMMA tiles.  K staged 32 at a time into LDS via
// global_load_async_to_lds_b128, double-buffered: tile k+1 DMA overlaps the
// WMMA burst on tile k (per-wave: 4 async issues/tile, s_wait_asynccnt 4).
// mode 0: bf16 out to (B,H,T,HD) | mode 1: bf16 out to (B,H,HD,T) | mode 2: f32 out
// ---------------------------------------------------------------------------
__global__ void __launch_bounds__(256)
gemm_xwt_kernel(const bf16* __restrict__ A, const bf16* __restrict__ W,
                void* __restrict__ Out, int mode) {
  extern __shared__ char smem[];
  bf16* lds = (bf16*)smem;  // 2 buffers x (As[128][LDA_] + Ws[128][LDA_])

  const int tid  = threadIdx.x;
  const int nb   = blockIdx.x;     // N block (0..7)
  const int mb   = blockIdx.y;     // M block (0..31)
  const int w    = tid >> 5;
  const int lane = tid & 31;
  const int hlf  = lane >> 4;
  const int l    = lane & 15;
  const int wm   = w >> 1;         // 0..3
  const int wn   = w & 1;          // 0..1

  // async staging: 128 rows x 32 cols bf16 = 512 x 16B chunks per matrix;
  // thread handles chunks {tid, tid+256} for A and for W.
  const int r0 = tid >> 2, q0 = tid & 3;               // chunk tid
  const int r1 = (tid + 256) >> 2, q1 = tid & 3;       // chunk tid+256
  const bf16* Arow0 = A + (size_t)(mb * 128 + r0) * DIM_ + q0 * 8;
  const bf16* Arow1 = A + (size_t)(mb * 128 + r1) * DIM_ + q1 * 8;
  const bf16* Wrow0 = W + (size_t)(nb * 128 + r0) * DIM_ + q0 * 8;
  const bf16* Wrow1 = W + (size_t)(nb * 128 + r1) * DIM_ + q1 * 8;

  auto issue_tile = [&](int k0, int buf) {
    bf16* As = lds + buf * (2 * 128 * LDA_);
    bf16* Ws = As + 128 * LDA_;
    async_copy_b128(As + r0 * LDA_ + q0 * 8, Arow0 + k0);
    async_copy_b128(As + r1 * LDA_ + q1 * 8, Arow1 + k0);
    async_copy_b128(Ws + r0 * LDA_ + q0 * 8, Wrow0 + k0);
    async_copy_b128(Ws + r1 * LDA_ + q1 * 8, Wrow1 + k0);
  };

  v8f acc[2][4];
#pragma unroll
  for (int mt = 0; mt < 2; ++mt)
#pragma unroll
    for (int nt = 0; nt < 4; ++nt) acc[mt][nt] = v8f{};

  issue_tile(0, 0);  // prologue

  const int KT = DIM_ / 32;  // 32 k-tiles
  for (int kt = 0; kt < KT; ++kt) {
    const int cur = kt & 1;
    if (kt + 1 < KT) {
      issue_tile((kt + 1) * 32, 1 - cur);  // overlap next-tile DMA with compute
      WAIT_ASYNCCNT(0x4);                  // current tile (issued first) complete
    } else {
      WAIT_ASYNCCNT(0x0);
    }
    __syncthreads();  // all waves' portions visible

    bf16* As = lds + cur * (2 * 128 * LDA_);
    bf16* Ws = As + 128 * LDA_;

    // A frags: lane holds row; element e -> k = (e&7) + 8*hlf + 16*(e>>3)
    v16bf afr[2];
#pragma unroll
    for (int mt = 0; mt < 2; ++mt) {
      int row = wm * 32 + mt * 16 + l;
#pragma unroll
      for (int e = 0; e < 16; ++e) {
        int kc = (e & 7) + 8 * hlf + 16 * (e >> 3);
        afr[mt][e] = As[row * LDA_ + kc];
      }
    }
    // B frags: lane holds col; element e -> k = (e&7) + 8*hlf + 16*(e>>3)
    v16bf bfr[4];
#pragma unroll
    for (int nt = 0; nt < 4; ++nt) {
      int cn = wn * 64 + nt * 16 + l;
#pragma unroll
      for (int e = 0; e < 16; ++e) bfr[nt][e] = Ws[cn * LDA_ + (e & 7) + 8 * hlf + 16 * (e >> 3)];
    }
#pragma unroll
    for (int mt = 0; mt < 2; ++mt)
#pragma unroll
      for (int nt = 0; nt < 4; ++nt)
        acc[mt][nt] = WMMA_BF16(afr[mt], bfr[nt], acc[mt][nt]);
    __syncthreads();  // all waves done reading before next issue overwrites
  }

  // Epilogue. D layout: vgpr v -> m = v + 8*hlf, n = l.
#pragma unroll
  for (int mt = 0; mt < 2; ++mt) {
#pragma unroll
    for (int nt = 0; nt < 4; ++nt) {
#pragma unroll
      for (int v = 0; v < 8; ++v) {
        int gi = mb * 128 + wm * 32 + mt * 16 + v + 8 * hlf;
        int gf = nb * 128 + wn * 64 + nt * 16 + l;
        float val = acc[mt][nt][v];
        if (mode == 2) {
          ((float*)Out)[(size_t)gi * DIM_ + gf] = val;
        } else {
          int b = gi >> 11, t = gi & (T_ - 1);
          int h = gf >> 6, d = gf & (HD_ - 1);
          size_t off = (mode == 0)
                           ? ((size_t)((b * H_ + h) * T_ + t)) * HD_ + d
                           : ((size_t)((b * H_ + h) * HD_ + d)) * T_ + t;
          ((bf16*)Out)[off] = (bf16)val;
        }
      }
    }
  }
}

// ---------------------------------------------------------------------------
// Fused attention: one block per (b, h, 32 query rows).  32x2048 f32 score
// tile in 256KB dynamic LDS (CDNA5 320KB WGP LDS).  Causal tiles skipped,
// wave32 shfl softmax, non-temporal att stream, P@V via WMMA from LDS probs.
// ---------------------------------------------------------------------------
__global__ void __launch_bounds__(256)
attn_kernel(const bf16* __restrict__ qh, const bf16* __restrict__ kh,
            const bf16* __restrict__ vT, float* __restrict__ att,
            bf16* __restrict__ yh) {
  extern __shared__ char smem[];
  float* S  = (float*)smem;            // [32][T_]
  bf16*  qs = (bf16*)(S + 32 * T_);    // [32][HD_]

  const int tid  = threadIdx.x;
  const int w    = tid >> 5;
  const int lane = tid & 31;
  const int hlf  = lane >> 4;
  const int l    = lane & 15;
  const int i0   = blockIdx.x * 32;
  const int h    = blockIdx.y;
  const int b    = blockIdx.z;
  const size_t bh = (size_t)(b * H_ + h);

  {
    const bf16* qbase = qh + (bh * T_ + i0) * HD_;
#pragma unroll
    for (int e = 0; e < 8; ++e) qs[tid + e * 256] = qbase[tid + e * 256];
  }
  __syncthreads();

  v16bf aq[2][2];
#pragma unroll
  for (int mt = 0; mt < 2; ++mt)
#pragma unroll
    for (int kk = 0; kk < 2; ++kk)
#pragma unroll
      for (int e = 0; e < 16; ++e) {
        int d = (e & 7) + 8 * hlf + 16 * (e >> 3) + 32 * kk;
        aq[mt][kk][e] = qs[(mt * 16 + l) * HD_ + d];
      }

  const int jt_max = i0 / 16 + 1;
  for (int jt = w; jt <= jt_max; jt += 8) {
    int j0 = jt * 16;
    v16bf bk[2];
    const bf16* krow = kh + (bh * T_ + j0 + l) * HD_;
#pragma unroll
    for (int kk = 0; kk < 2; ++kk)
#pragma unroll
      for (int e = 0; e < 16; ++e) bk[kk][e] = krow[(e & 7) + 8 * hlf + 16 * (e >> 3) + 32 * kk];
#pragma unroll
    for (int mt = 0; mt < 2; ++mt) {
      v8f c = {};
      c = WMMA_BF16(aq[mt][0], bk[0], c);
      c = WMMA_BF16(aq[mt][1], bk[1], c);
#pragma unroll
      for (int v = 0; v < 8; ++v) {
        int i = i0 + mt * 16 + v + 8 * hlf;
        int j = j0 + l;
        float s = c[v] * SCALE_;
        if (j > i) s = -__builtin_inff();
        S[(i - i0) * T_ + j] = s;
      }
    }
  }
  __syncthreads();

  const int jcomp = i0 + 32;
  for (int r = w; r < 32; r += 8) {
    int i = i0 + r;
    float* row = S + r * T_;
    float m = -__builtin_inff();
    for (int j = lane; j < jcomp; j += 32) m = fmaxf(m, row[j]);
#pragma unroll
    for (int off = 16; off > 0; off >>= 1) m = fmaxf(m, __shfl_xor(m, off));
    float sum = 0.0f;
    for (int j = lane; j < jcomp; j += 32) {
      float p = __expf(row[j] - m);
      row[j] = p;
      sum += p;
    }
#pragma unroll
    for (int off = 16; off > 0; off >>= 1) sum += __shfl_xor(sum, off);
    float inv = 1.0f / sum;
    float* out = att + (bh * T_ + i) * T_;
    for (int j = lane; j < jcomp; j += 32) {
      float p = row[j] * inv;
      row[j] = p;
      __builtin_nontemporal_store(p, out + j);
    }
    for (int j = jcomp + lane; j < T_; j += 32)
      __builtin_nontemporal_store(0.0f, out + j);
  }
  __syncthreads();

  const int mt = w & 1;
  const int nt = w >> 1;
  v8f acc = {};
  const bf16* vbase = vT + (bh * HD_ + nt * 16 + l) * T_;
  const int njc = jcomp >> 5;
  for (int jc = 0; jc < njc; ++jc) {
    v16bf ap, bv;
#pragma unroll
    for (int e = 0; e < 16; ++e) {
      int j = jc * 32 + (e & 7) + 8 * hlf + 16 * (e >> 3);
      ap[e] = (bf16)S[(mt * 16 + l) * T_ + j];
    }
#pragma unroll
    for (int e = 0; e < 16; ++e) bv[e] = vbase[jc * 32 + (e & 7) + 8 * hlf + 16 * (e >> 3)];
    acc = WMMA_BF16(ap, bv, acc);
  }
#pragma unroll
  for (int v = 0; v < 8; ++v) {
    int t = i0 + mt * 16 + v + 8 * hlf;
    int f = h * HD_ + nt * 16 + l;
    yh[((size_t)(b * T_ + t)) * DIM_ + f] = (bf16)acc[v];
  }
}

// ---------------------------------------------------------------------------
extern "C" void kernel_launch(void* const* d_in, const int* in_sizes, int n_in,
                              void* d_out, int out_size, void* d_ws, size_t ws_size,
                              hipStream_t stream) {
  (void)in_sizes; (void)n_in; (void)out_size; (void)ws_size;
  const float* Q  = (const float*)d_in[0];
  const float* K  = (const float*)d_in[1];
  const float* V  = (const float*)d_in[2];
  const float* Wq = (const float*)d_in[3];
  const float* Wk = (const float*)d_in[4];
  const float* Wv = (const float*)d_in[5];
  const float* Wo = (const float*)d_in[6];

  float* y   = (float*)d_out;                   // (B,T,DIM) f32
  float* att = y + (size_t)B_ * T_ * DIM_;      // (B,H,T,T) f32

  const size_t NACT = (size_t)B_ * T_ * DIM_;   // 4M elements
  const size_t NW   = (size_t)DIM_ * DIM_;      // 1M elements
  bf16* Qb  = (bf16*)d_ws;                      // bf16 copies for async-DMA GEMMs
  bf16* Kb  = Qb + NACT;
  bf16* Vb  = Kb + NACT;
  bf16* Wqb = Vb + NACT;
  bf16* Wkb = Wqb + NW;
  bf16* Wvb = Wkb + NW;
  bf16* Wob = Wvb + NW;
  bf16* qh  = Wob + NW;                         // (B,H,T,HD)
  bf16* kh  = qh + NACT;                        // (B,H,T,HD)
  bf16* vT  = kh + NACT;                        // (B,H,HD,T)
  bf16* yh  = vT + NACT;                        // (B,T,DIM)

  dim3 cblk(256);
  cvt_f32_bf16<<<dim3(NACT / 1024), cblk, 0, stream>>>(Q, Qb);
  cvt_f32_bf16<<<dim3(NACT / 1024), cblk, 0, stream>>>(K, Kb);
  cvt_f32_bf16<<<dim3(NACT / 1024), cblk, 0, stream>>>(V, Vb);
  cvt_f32_bf16<<<dim3(NW / 1024), cblk, 0, stream>>>(Wq, Wqb);
  cvt_f32_bf16<<<dim3(NW / 1024), cblk, 0, stream>>>(Wk, Wkb);
  cvt_f32_bf16<<<dim3(NW / 1024), cblk, 0, stream>>>(Wv, Wvb);
  cvt_f32_bf16<<<dim3(NW / 1024), cblk, 0, stream>>>(Wo, Wob);

  dim3 gblk(256);
  dim3 ggrid(DIM_ / 128, (B_ * T_) / 128);      // (8, 32)
  size_t gemm_lds = (size_t)2 * 2 * 128 * LDA_ * sizeof(bf16);  // double-buffered

  gemm_xwt_kernel<<<ggrid, gblk, gemm_lds, stream>>>(Qb, Wqb, qh, 0);
  gemm_xwt_kernel<<<ggrid, gblk, gemm_lds, stream>>>(Kb, Wkb, kh, 0);
  gemm_xwt_kernel<<<ggrid, gblk, gemm_lds, stream>>>(Vb, Wvb, vT, 1);

  size_t attn_lds = (size_t)32 * T_ * sizeof(float) + (size_t)32 * HD_ * sizeof(bf16);
  attn_kernel<<<dim3(T_ / 32, H_, B_), gblk, attn_lds, stream>>>(qh, kh, vT, att, yh);

  gemm_xwt_kernel<<<ggrid, gblk, gemm_lds, stream>>>(yh, Wob, y, 2);
}
